// BeltramiTransformerBlock_74423193305342
// MI455X (gfx1250) — matrix operand order repaired and checked by reading, not compiled, not run on hardware
//
#include <hip/hip_runtime.h>
#include <math.h>

// ---------------------------------------------------------------------------
// Types
// ---------------------------------------------------------------------------
typedef __bf16 bf16_t;
typedef bf16_t v16bf __attribute__((ext_vector_type(16)));
typedef float  v8f   __attribute__((ext_vector_type(8)));
typedef unsigned int  u32x4 __attribute__((ext_vector_type(4)));
typedef int           i32x8 __attribute__((ext_vector_type(8)));
typedef int           i32x4 __attribute__((ext_vector_type(4)));
typedef unsigned short bfs;   // bf16 storage type for LDS/global staging

#define WANT_TDM 1
#if WANT_TDM && defined(__gfx1250__) && __has_builtin(__builtin_amdgcn_tensor_load_to_lds)
#define USE_TDM 1
#else
#define USE_TDM 0
#endif

// Problem constants
#define Bsz 4
#define DIM 192
#define Hh 256
#define Ww 256
#define HEADS 6
#define WS 8
#define Kbpe 32
#define NWIN 4096          // B * (H/8) * (W/8)
#define NTOK 64            // WS*WS
#define NPIX 262144        // B*H*W

// ---------------------------------------------------------------------------
// bf16 helpers: hardware RNE convert (v_cvt_bf16_f32) instead of a 5-op
// software rounding sequence.
// ---------------------------------------------------------------------------
__device__ __forceinline__ bfs f2bf(float f) {
  bf16_t h = (bf16_t)f;
  bfs r;
  __builtin_memcpy(&r, &h, 2);
  return r;
}

__device__ __forceinline__ v8f wmma_bf16(v16bf a, v16bf b, v8f c) {
#if defined(__gfx1250__)
  return __builtin_amdgcn_wmma_f32_16x16x32_bf16(false, a, false, b, (short)0, c,
                                                 false, false);
#else
  return c;  // host-pass placeholder (never codegen'd for host)
#endif
}

// ---------------------------------------------------------------------------
// Fragment loaders.
// A 16x32 bf16 fragment: lane m=l&15, half=l>>4; VGPR j holds
// K = 2j + 16*half  (pairs K,K+1 -> one dword).
// B 32x16 bf16 fragment (same k order as A): lane n=l&15, half=l>>4; VGPR j
// holds K = 2j + 16*half (pairs contiguous).
// Source arrays hold the "transposed" matrix row-major so pairs are dwords.
// ---------------------------------------------------------------------------
__device__ __forceinline__ v16bf ldAT(const bfs* __restrict__ base, int ld, int lane) {
  const int m = lane & 15, half = lane >> 4;
  union { v16bf v; unsigned u[8]; } r;
#pragma unroll
  for (int j = 0; j < 8; ++j) {
    const int k = 2 * j + 16 * half;
    r.u[j] = *(const unsigned*)(base + m * ld + k);
  }
  return r.v;
}

__device__ __forceinline__ v16bf ldBT(const bfs* __restrict__ base, int ld, int lane) {
  const int n = lane & 15, half = lane >> 4;
  union { v16bf v; unsigned u[8]; } r;
#pragma unroll
  for (int j = 0; j < 8; ++j) {
    const int k = 2 * j + 16 * half;
    r.u[j] = *(const unsigned*)(base + n * ld + k);
  }
  return r.v;
}

// Pre-swizzled weight tile: 32 bytes contiguous per lane -> 2x b128 loads.
__device__ __forceinline__ v16bf ldBW(const bfs* __restrict__ tile, int lane) {
  return *(const v16bf*)(tile + lane * 16);
}

// A fragment from TDM-padded LDS image (+1 dword pad after every 32 dwords).
__device__ __forceinline__ v16bf ldA_pad(const unsigned* __restrict__ dw, int row0,
                                         int ld_dw, int kdw0, int lane) {
  const int m = lane & 15, half = lane >> 4;
  union { v16bf v; unsigned u[8]; } r;
#pragma unroll
  for (int j = 0; j < 8; ++j) {
    const int k = 2 * j + 16 * half;
    const int d = (row0 + m) * ld_dw + kdw0 + (k >> 1);
    r.u[j] = dw[d + (d >> 5)];
  }
  return r.v;
}

// ---------------------------------------------------------------------------
// TDM 1-D copy global->LDS with LDS padding (1 dword per 32 dwords).
// D# built per CDNA5 ISA.
// This toolchain exposes the 6-arg builtin:
//   (uint32x4 g0, int32x8 g1, int32x4 g2, int32x4 g3, int32x8 g4, i32 cpol)
// ---------------------------------------------------------------------------
#if USE_TDM
__device__ __forceinline__ void tdm_load_1d_pad(unsigned lds_addr, const void* gptr,
                                                unsigned nelem /*2B units*/) {
  const unsigned long long ga = (unsigned long long)gptr;
  u32x4 g0;
  g0.x = 1u;                                            // count=1, user mode
  g0.y = lds_addr;                                      // lds_addr [63:32]
  g0.z = (unsigned)(ga & 0xffffffffu);                  // global_addr lo
  g0.w = (unsigned)((ga >> 32) & 0x1ffffffu) | (2u << 30);  // addr hi | type=2
  i32x8 g1 = (i32x8)0;
  g1[0] = (int)((1u << 16)            // data_size = 2 bytes
              | (1u << 20)            // pad_enable
              | (4u << 22));          // pad_interval = 32 dwords (+1 dword pad)
  g1[1] = (int)((nelem & 0xffffu) << 16);               // tensor_dim0 lo16
  g1[2] = (int)(((nelem >> 16) & 0xffffu) | (1u << 16)); // td0 hi16 | td1=1 lo
  g1[3] = (int)((nelem & 0xffffu) << 16);               // tile_dim0
  g1[4] = 1;                                            // tile_dim1 = 1
  g1[5] = (int)nelem;                                   // tensor_dim0_stride
  i32x4 z4 = (i32x4)0;
  i32x8 z8 = (i32x8)0;
  __builtin_amdgcn_tensor_load_to_lds(g0, g1, z4, z4, z8, 0);
}
#endif

// ---------------------------------------------------------------------------
// K0: weight f32 -> bf16 swizzled into B-fragment tile order.
// out[((nt*KT + kt)*32 + lane)*16 + i] = W[kt*32 + i + 16*(lane>>4)][nt*16 + (lane&15)]
// ---------------------------------------------------------------------------
__global__ void __launch_bounds__(256) k_swizzle_w(const float* __restrict__ W,
                                                   bfs* __restrict__ out,
                                                   int KD, int ND) {
  const int idx = blockIdx.x * 256 + threadIdx.x;
  if (idx >= KD * ND) return;
  const int i = idx & 15;
  const int lane = (idx >> 4) & 31;
  const int tile = idx >> 9;
  const int KT = KD >> 5;
  const int kt = tile % KT, nt = tile / KT;
  const int k = i + 16 * (lane >> 4);
  const int n = lane & 15;
  out[idx] = f2bf(W[(size_t)(kt * 32 + k) * ND + (nt * 16 + n)]);
}

// ---------------------------------------------------------------------------
// K1: transpose NCHW->window-major, LayerNorm1, bf16 pack; also window the BPE.
// ---------------------------------------------------------------------------
__global__ void __launch_bounds__(256) k_ln1_window(const float* __restrict__ x,
                                                    const float* __restrict__ bpe,
                                                    const float* __restrict__ w,
                                                    const float* __restrict__ bvec,
                                                    bfs* __restrict__ xnw,
                                                    bfs* __restrict__ bww) {
  const int pid = blockIdx.x * 256 + threadIdx.x;   // over B*H*W
  const int b = pid >> 16, h = (pid >> 8) & 255, wx = pid & 255;
  const int wi = ((b << 5) + (h >> 3)) * 32 + (wx >> 3);
  const int t = ((h & 7) << 3) + (wx & 7);
  const float* px = x + (size_t)b * DIM * 65536 + h * 256 + wx;
  float s = 0.f, s2 = 0.f;
#pragma unroll 4
  for (int c = 0; c < DIM; ++c) { float v = px[(size_t)c * 65536]; s += v; s2 += v * v; }
  const float mean = s * (1.f / DIM);
  const float var = s2 * (1.f / DIM) - mean * mean;
  const float rstd = rsqrtf(var + 1e-5f);
  bfs* q = xnw + ((size_t)wi * NTOK + t) * DIM;
#pragma unroll 4
  for (int c = 0; c < DIM; ++c) {
    float v = px[(size_t)c * 65536];
    q[c] = f2bf((v - mean) * rstd * w[c] + bvec[c]);
  }
  const float* pb = bpe + (size_t)b * Kbpe * 65536 + h * 256 + wx;
  bfs* qb = bww + ((size_t)wi * NTOK + t) * Kbpe;
#pragma unroll
  for (int c = 0; c < Kbpe; ++c) qb[c] = f2bf(pb[(size_t)c * 65536]);
}

// ---------------------------------------------------------------------------
// Kernel A: per-window fused QKV + BPE GEMMs, attention, proj, residual.
// 1 workgroup = 1 window, 128 threads = 4 waves, wave owns 16 token rows.
// ---------------------------------------------------------------------------
#define A_XN   0        // 6336 dw (padded 64x96dw)           25344 B
#define A_BW   25344    // 1056 dw (padded 64x16dw)            4224 B
#define A_QS   29568    // 64x198 bfs                         25344 B
#define A_KS   54912
#define A_BQS  80256
#define A_BKS  105600
#define A_VT   130944   // 6 heads x 32ch x 66 tok
#define A_SS   156288   // 64x65 f32                          16640 B
#define A_PS   172928   // 64x66 bfs                           8448 B
#define A_OS   181376   // 64x198 bfs
#define A_OF   206720   // 64x193 f32                         49408 B
#define A_SMEM 256128

__global__ void __launch_bounds__(128) k_attn(
    const float* __restrict__ x, const bfs* __restrict__ xnw, const bfs* __restrict__ bww,
    const bfs* __restrict__ qkv_sw, const bfs* __restrict__ bpe_sw,
    const bfs* __restrict__ proj_sw, const float* __restrict__ qkv_b,
    const float* __restrict__ bpe_b, const float* __restrict__ proj_b,
    float* __restrict__ xh2) {
  extern __shared__ __align__(16) char smem[];
  unsigned* xn_dw = (unsigned*)(smem + A_XN);
  unsigned* bw_dw = (unsigned*)(smem + A_BW);
  bfs* qs  = (bfs*)(smem + A_QS);
  bfs* ks  = (bfs*)(smem + A_KS);
  bfs* bqs = (bfs*)(smem + A_BQS);
  bfs* bks = (bfs*)(smem + A_BKS);
  bfs* vt  = (bfs*)(smem + A_VT);
  float* ss = (float*)(smem + A_SS);
  bfs* ps  = (bfs*)(smem + A_PS);
  bfs* os_ = (bfs*)(smem + A_OS);
  float* of = (float*)(smem + A_OF);

  const int tid = threadIdx.x, lane = tid & 31, wave = tid >> 5;
  const int wi = blockIdx.x;
  const int b = wi >> 10, wh = (wi >> 5) & 31, wwid = wi & 31;
  const int h0 = wh << 3, w0 = wwid << 3;
  const int row0 = wave << 4;
  const int n = lane & 15, mo = (lane >> 4) << 3;
  const v8f vz = (v8f)0.0f;

  // ---- stage window inputs into LDS (TDM async DMA, padded) ----
#if USE_TDM
  if (wave == 0) {
    tdm_load_1d_pad((unsigned)(unsigned long long)(void*)xn_dw,
                    xnw + (size_t)wi * (NTOK * DIM), NTOK * DIM);
    tdm_load_1d_pad((unsigned)(unsigned long long)(void*)bw_dw,
                    bww + (size_t)wi * (NTOK * Kbpe), NTOK * Kbpe);
    __builtin_amdgcn_s_wait_tensorcnt(0);
  }
#else
  {
    const unsigned* s1 = (const unsigned*)(xnw + (size_t)wi * (NTOK * DIM));
    for (int i = tid; i < 6144; i += 128) xn_dw[i + (i >> 5)] = s1[i];
    const unsigned* s2 = (const unsigned*)(bww + (size_t)wi * (NTOK * Kbpe));
    for (int i = tid; i < 1024; i += 128) bw_dw[i + (i >> 5)] = s2[i];
  }
#endif
  __syncthreads();

  // ---- QKV GEMM: 16x576 slice per wave;  q,k -> [tok][ch], v -> [ch][tok] ----
  for (int nt = 0; nt < 36; ++nt) {
    v8f acc = vz;
#pragma unroll
    for (int kt = 0; kt < 6; ++kt) {
      v16bf a  = ldA_pad(xn_dw, row0, 96, kt * 16, lane);
      v16bf bb = ldBW(qkv_sw + (nt * 6 + kt) * 512, lane);
      acc = wmma_bf16(a, bb, acc);
    }
    const int col16 = nt << 4;
    const float bias = qkv_b[col16 + n];
    const int which = col16 / DIM, cc0 = col16 % DIM;
    if (which == 0) {
#pragma unroll
      for (int r = 0; r < 8; ++r) qs[(row0 + mo + r) * 198 + cc0 + n] = f2bf(acc[r] + bias);
    } else if (which == 1) {
#pragma unroll
      for (int r = 0; r < 8; ++r) ks[(row0 + mo + r) * 198 + cc0 + n] = f2bf(acc[r] + bias);
    } else {
      const int cc = cc0 + n, hh = cc >> 5, c5 = cc & 31;
#pragma unroll
      for (int r = 0; r < 8; ++r) vt[(hh * 32 + c5) * 66 + row0 + mo + r] = f2bf(acc[r] + bias);
    }
  }
  // ---- BPE GEMM: bq,bk (K=32 -> single WMMA per tile) ----
  {
    v16bf a = ldA_pad(bw_dw, row0, 16, 0, lane);
    for (int nt = 0; nt < 24; ++nt) {
      v16bf bb = ldBW(bpe_sw + nt * 512, lane);
      v8f acc = wmma_bf16(a, bb, vz);
      const int col16 = nt << 4;
      const float bias = bpe_b[col16 + n];
      bfs* dst = (col16 < DIM) ? bqs : bks;
      const int cc0 = col16 % DIM;
#pragma unroll
      for (int r = 0; r < 8; ++r) dst[(row0 + mo + r) * 198 + cc0 + n] = f2bf(acc[r] + bias);
    }
  }
  __syncthreads();   // k, bk, v needed by all waves

  // ---- attention, per head (wave-local rows) ----
  const float scale = 0.17677669529663687f;  // 1/sqrt(32)
  for (int hh = 0; hh < 6; ++hh) {
    v16bf aq  = ldAT(qs  + row0 * 198 + hh * 32, 198, lane);
    v16bf abq = ldAT(bqs + row0 * 198 + hh * 32, 198, lane);
#pragma unroll
    for (int nt = 0; nt < 4; ++nt) {
      v16bf bk  = ldBT(ks  + (nt * 16) * 198 + hh * 32, 198, lane);
      v8f s = wmma_bf16(aq, bk, vz);
      v16bf bbk = ldBT(bks + (nt * 16) * 198 + hh * 32, 198, lane);
      s = wmma_bf16(abq, bbk, s);
#pragma unroll
      for (int r = 0; r < 8; ++r) ss[(row0 + mo + r) * 65 + nt * 16 + n] = s[r] * scale;
    }
    // softmax: 2 lanes per row (halves combined with shfl_xor 16)
    {
      const int rr = row0 + (lane & 15);
      const int c0 = (lane >> 4) << 5;
      float mx = -1e30f;
#pragma unroll
      for (int c = 0; c < 32; ++c) mx = fmaxf(mx, ss[rr * 65 + c0 + c]);
      mx = fmaxf(mx, __shfl_xor(mx, 16, 32));
      float sum = 0.f;
#pragma unroll
      for (int c = 0; c < 32; ++c) sum += __expf(ss[rr * 65 + c0 + c] - mx);
      sum += __shfl_xor(sum, 16, 32);
      const float inv = 1.0f / sum;
#pragma unroll
      for (int c = 0; c < 32; ++c)
        ps[rr * 66 + c0 + c] = f2bf(__expf(ss[rr * 65 + c0 + c] - mx) * inv);
    }
    // O = P @ V
#pragma unroll
    for (int nt2 = 0; nt2 < 2; ++nt2) {
      v8f o = vz;
#pragma unroll
      for (int kt = 0; kt < 2; ++kt) {
        v16bf ap = ldAT(ps + row0 * 66 + kt * 32, 66, lane);
        v16bf bv = ldBT(vt + (hh * 32 + nt2 * 16) * 66 + kt * 32, 66, lane);
        o = wmma_bf16(ap, bv, o);
      }
#pragma unroll
      for (int r = 0; r < 8; ++r)
        os_[(row0 + mo + r) * 198 + hh * 32 + nt2 * 16 + n] = f2bf(o[r]);
    }
  }

  // ---- projection GEMM + bias -> LDS staging ----
  for (int nt = 0; nt < 12; ++nt) {
    v8f acc = vz;
#pragma unroll
    for (int kt = 0; kt < 6; ++kt) {
      v16bf a  = ldAT(os_ + row0 * 198 + kt * 32, 198, lane);
      v16bf bb = ldBW(proj_sw + (nt * 6 + kt) * 512, lane);
      acc = wmma_bf16(a, bb, acc);
    }
    const float bias = proj_b[nt * 16 + n];
#pragma unroll
    for (int r = 0; r < 8; ++r) of[(row0 + mo + r) * 193 + nt * 16 + n] = acc[r] + bias;
  }
  // ---- shortcut (gather from NCHW x) + coalesced NHWC store ----
  for (int j = lane; j < 16 * DIM; j += 32) {
    const int tl = j / DIM, c = j - tl * DIM;
    const int t = row0 + tl;
    const int hy = h0 + (t >> 3), wx2 = w0 + (t & 7);
    const float sc = x[(((size_t)b * DIM + c) << 16) + hy * 256 + wx2];
    xh2[((size_t)((b << 16) + hy * 256 + wx2)) * DIM + c] = of[t * 193 + c] + sc;
  }
}

// ---------------------------------------------------------------------------
// Kernel B: LN2 + FC1 + GELU + FC2 + residual + NHWC->NCHW store.
// 1 workgroup = 64 consecutive pixels (one quarter image row).
// ---------------------------------------------------------------------------
#define B_X2  0         // 64x193 f32   49408
#define B_AS  49408     // 64x198 bfs   25344
#define B_HS  74752     // 64x774 bfs   99072
#define B_OF  173824    // 64x193 f32   49408
#define B_SMEM 223232

__global__ void __launch_bounds__(128) k_mlp(
    const float* __restrict__ xh2, const float* __restrict__ ln2w,
    const float* __restrict__ ln2b, const bfs* __restrict__ fc1_sw,
    const float* __restrict__ fc1_b, const bfs* __restrict__ fc2_sw,
    const float* __restrict__ fc2_b, float* __restrict__ out) {
  extern __shared__ __align__(16) char smem[];
  float* x2s = (float*)(smem + B_X2);
  bfs*   a_s = (bfs*)(smem + B_AS);
  bfs*   h_s = (bfs*)(smem + B_HS);
  float* of  = (float*)(smem + B_OF);
  const int tid = threadIdx.x, lane = tid & 31, wave = tid >> 5;
  const int row0 = wave << 4, n = lane & 15, mo = (lane >> 4) << 3;
  const int p0 = blockIdx.x << 6;
  const int b = p0 >> 16, hp = (p0 >> 8) & 255, w0 = p0 & 255;
  const v8f vz = (v8f)0.0f;

  const float* src = xh2 + (size_t)p0 * DIM;
  for (int i = tid; i < 64 * DIM; i += 128) {
    const int t = i / DIM, c = i - t * DIM;
    x2s[t * 193 + c] = src[i];
  }
  __syncthreads();
  if (tid < 64) {
    const int t = tid;
    float s = 0.f, s2 = 0.f;
#pragma unroll 4
    for (int c = 0; c < DIM; ++c) { float v = x2s[t * 193 + c]; s += v; s2 += v * v; }
    const float mean = s * (1.f / DIM);
    const float var = s2 * (1.f / DIM) - mean * mean;
    const float rstd = rsqrtf(var + 1e-5f);
#pragma unroll 4
    for (int c = 0; c < DIM; ++c) {
      float v = x2s[t * 193 + c];
      a_s[t * 198 + c] = f2bf((v - mean) * rstd * ln2w[c] + ln2b[c]);
    }
  }
  __syncthreads();

  // FC1 + exact GELU
  for (int nt = 0; nt < 48; ++nt) {
    v8f acc = vz;
#pragma unroll
    for (int kt = 0; kt < 6; ++kt) {
      v16bf a  = ldAT(a_s + row0 * 198 + kt * 32, 198, lane);
      v16bf bb = ldBW(fc1_sw + (nt * 6 + kt) * 512, lane);
      acc = wmma_bf16(a, bb, acc);
    }
    const float bias = fc1_b[nt * 16 + n];
#pragma unroll
    for (int r = 0; r < 8; ++r) {
      float g = acc[r] + bias;
      g = 0.5f * g * (1.0f + erff(g * 0.70710678118654752f));
      h_s[(row0 + mo + r) * 774 + nt * 16 + n] = f2bf(g);
    }
  }
  // FC2 + residual (wave-local rows -> no barrier needed after FC1)
  for (int nt = 0; nt < 12; ++nt) {
    v8f acc = vz;
#pragma unroll
    for (int kt = 0; kt < 24; ++kt) {
      v16bf a  = ldAT(h_s + row0 * 774 + kt * 32, 774, lane);
      v16bf bb = ldBW(fc2_sw + (nt * 24 + kt) * 512, lane);
      acc = wmma_bf16(a, bb, acc);
    }
    const float bias = fc2_b[nt * 16 + n];
#pragma unroll
    for (int r = 0; r < 8; ++r) {
      const int t = row0 + mo + r;
      of[t * 193 + nt * 16 + n] = acc[r] + bias + x2s[t * 193 + nt * 16 + n];
    }
  }
  __syncthreads();
  // coalesced NCHW store: lanes sweep w, fixed channel per 64-elem group
  for (int i = tid; i < 64 * DIM; i += 128) {
    const int c = i >> 6, t = i & 63;
    out[(((size_t)b * DIM + c) << 16) + hp * 256 + w0 + t] = of[t * 193 + c];
  }
}

// ---------------------------------------------------------------------------
// Host launcher
// ---------------------------------------------------------------------------
extern "C" void kernel_launch(void* const* d_in, const int* in_sizes, int n_in,
                              void* d_out, int out_size, void* d_ws, size_t ws_size,
                              hipStream_t stream) {
  (void)in_sizes; (void)n_in; (void)out_size; (void)ws_size;
  const float* x     = (const float*)d_in[0];
  const float* bpe   = (const float*)d_in[1];
  const float* ln1w  = (const float*)d_in[2];
  const float* ln1b  = (const float*)d_in[3];
  const float* qkvw  = (const float*)d_in[4];
  const float* qkvb  = (const float*)d_in[5];
  const float* bpew  = (const float*)d_in[6];
  const float* bpeb  = (const float*)d_in[7];
  const float* projw = (const float*)d_in[8];
  const float* projb = (const float*)d_in[9];
  const float* ln2w  = (const float*)d_in[10];
  const float* ln2b  = (const float*)d_in[11];
  const float* fc1w  = (const float*)d_in[12];
  const float* fc1b  = (const float*)d_in[13];
  const float* fc2w  = (const float*)d_in[14];
  const float* fc2b  = (const float*)d_in[15];
  float* out = (float*)d_out;

  // workspace carve (all 256B aligned)
  char* ws = (char*)d_ws;
  bfs* qkv_sw = (bfs*)(ws + 0);          // 192*576*2  = 221184
  bfs* bpe_sw = (bfs*)(ws + 221184);     //  32*384*2  =  24576
  bfs* proj_sw= (bfs*)(ws + 245760);     // 192*192*2  =  73728
  bfs* fc1_sw = (bfs*)(ws + 319488);     // 192*768*2  = 294912
  bfs* fc2_sw = (bfs*)(ws + 614400);     // 768*192*2  = 294912
  bfs* xnw    = (bfs*)(ws + 909312);     // 4096*64*192*2 = 100663296
  bfs* bww    = (bfs*)(ws + 101572608);  // 4096*64*32*2  =  16777216
  float* xh2  = (float*)(ws + 118349824);// 262144*192*4  = 201326592

  (void)hipFuncSetAttribute((const void*)k_attn,
                            hipFuncAttributeMaxDynamicSharedMemorySize, A_SMEM);
  (void)hipFuncSetAttribute((const void*)k_mlp,
                            hipFuncAttributeMaxDynamicSharedMemorySize, B_SMEM);

  // weight swizzles (bf16, B-fragment tile order)
  k_swizzle_w<<<432, 256, 0, stream>>>(qkvw, qkv_sw, 192, 576);
  k_swizzle_w<<<48,  256, 0, stream>>>(bpew, bpe_sw, 32, 384);
  k_swizzle_w<<<144, 256, 0, stream>>>(projw, proj_sw, 192, 192);
  k_swizzle_w<<<576, 256, 0, stream>>>(fc1w, fc1_sw, 192, 768);
  k_swizzle_w<<<576, 256, 0, stream>>>(fc2w, fc2_sw, 768, 192);

  k_ln1_window<<<NPIX / 256, 256, 0, stream>>>(x, bpe, ln1w, ln1b, xnw, bww);

  k_attn<<<NWIN, 128, A_SMEM, stream>>>(x, xnw, bww, qkv_sw, bpe_sw, proj_sw,
                                        qkvb, bpeb, projb, xh2);

  k_mlp<<<NPIX / 64, 128, B_SMEM, stream>>>(xh2, ln2w, ln2b, fc1_sw, fc1b,
                                            fc2_sw, fc2b, out);
}
